// ABMILAggregator_45938970198547
// MI455X (gfx1250) — compile-verified, matrix operand order checked by reading, not run on hardware
//
#include <hip/hip_runtime.h>
#include <hip/hip_bf16.h>
#include <math.h>

// Problem constants (match reference)
constexpr int kN = 262144;   // rows
constexpr int kD = 1024;     // feature dim
constexpr int kH = 256;      // hidden dim
constexpr int kB = 16;       // segments

typedef __bf16 bf16;
typedef bf16  v16bf __attribute__((ext_vector_type(16)));
typedef bf16  v2bf  __attribute__((ext_vector_type(2)));
typedef float v8f   __attribute__((ext_vector_type(8)));

// ---------------------------------------------------------------------------
// Kernel 0: convert Wv/Wu (D x H, f32, row-major) into transposed bf16
// (H x D) so B-fragment loads are K-contiguous 4-byte loads.
// ---------------------------------------------------------------------------
__global__ __launch_bounds__(256) void convert_kernel(
    const float* __restrict__ Wv, const float* __restrict__ Wu,
    bf16* __restrict__ WvT, bf16* __restrict__ WuT)
{
    int idx = blockIdx.x * 256 + threadIdx.x;      // over H*D
    if (idx < kH * kD) {
        int n = idx / kD;
        int k = idx - n * kD;
        WvT[idx] = (bf16)Wv[k * kH + n];
        WuT[idx] = (bf16)Wu[k * kH + n];
    }
}

// ---------------------------------------------------------------------------
// Kernel 1: zero the (B, D) slide_features accumulator in d_out.
// ---------------------------------------------------------------------------
__global__ __launch_bounds__(256) void zero_kernel(float* __restrict__ p)
{
    p[blockIdx.x * 256 + threadIdx.x] = 0.0f;
}

// ---------------------------------------------------------------------------
// Kernel 2: fused gated-attention score.
//   A[i] = sum_n tanh(x_i.Wv_n + bv_n) * sigmoid(x_i.Wu_n + bu_n) * w_n
// One wave per 16-row tile. bf16 WMMA 16x16x32, D/32 K-steps,
// 16 N-tiles done as 4 chunks of 4 (A-fragment reused 4x per build).
// ---------------------------------------------------------------------------
__global__ __launch_bounds__(256) void score_kernel(
    const float* __restrict__ x,
    const bf16*  __restrict__ WvT, const bf16* __restrict__ WuT,
    const float* __restrict__ bv,  const float* __restrict__ bu,
    const float* __restrict__ w,
    float* __restrict__ A_out)
{
    const int lane = threadIdx.x & 31;
    const int wave = threadIdx.x >> 5;
    const int tile = blockIdx.x * 8 + wave;
    const int row_base = tile * 16;
    const int hl = lane >> 4;        // lane half: selects K-group (ISA 7.12.2)
    const int mn = lane & 15;        // A-frag row / B-frag & C column
    const size_t xrow = (size_t)(row_base + mn) * kD;

    float acc[8];
#pragma unroll
    for (int r = 0; r < 8; ++r) acc[r] = 0.0f;

    for (int nc = 0; nc < 4; ++nc) {            // 4 chunks of 4 N-tiles
        v8f cv[4] = {};
        v8f cu[4] = {};

        for (int kt = 0; kt < kD / 32; ++kt) {  // 32 K-steps
            const int k0 = kt * 32;

            // Build A fragment (16x32 bf16) from f32 x rows.
            v16bf a;
#pragma unroll
            for (int i = 0; i < 8; ++i) {
                const int k = k0 + hl * 8 + ((i < 4) ? 2 * i : 16 + 2 * (i - 4));
                const float2 xp = *(const float2*)(x + xrow + k);
                a[2 * i]     = (bf16)xp.x;
                a[2 * i + 1] = (bf16)xp.y;
            }

#pragma unroll
            for (int j = 0; j < 4; ++j) {
                const size_t wrow = (size_t)((nc * 4 + j) * 16 + mn) * kD;
                v16bf bvf, buf;
#pragma unroll
                for (int i = 0; i < 8; ++i) {
                    const int k = k0 + hl * 8 + ((i < 4) ? 2 * i : 16 + 2 * (i - 4));
                    const v2bf pv = *(const v2bf*)(WvT + wrow + k);
                    const v2bf pu = *(const v2bf*)(WuT + wrow + k);
                    bvf[2 * i] = pv.x; bvf[2 * i + 1] = pv.y;
                    buf[2 * i] = pu.x; buf[2 * i + 1] = pu.y;
                }
                cv[j] = __builtin_amdgcn_wmma_f32_16x16x32_bf16(
                    false, a, false, bvf, (short)0, cv[j], false, false);
                cu[j] = __builtin_amdgcn_wmma_f32_16x16x32_bf16(
                    false, a, false, buf, (short)0, cu[j], false, false);
            }
        }

        // Epilogue: bias + tanh*sigmoid gate, dot with w, accumulate per row.
#pragma unroll
        for (int j = 0; j < 4; ++j) {
            const int ncol = (nc * 4 + j) * 16 + mn;
            const float bvn = bv[ncol];
            const float bun = bu[ncol];
            const float wn  = w[ncol];
#pragma unroll
            for (int r = 0; r < 8; ++r) {
                const float t = tanhf(cv[j][r] + bvn);
                const float s = 1.0f / (1.0f + __expf(-(cu[j][r] + bun)));
                acc[r] += t * s * wn;
            }
        }
    }

    // Reduce the 16 column-lanes within each half-wave (rows m = r + 8*hl).
#pragma unroll
    for (int off = 1; off < 16; off <<= 1) {
#pragma unroll
        for (int r = 0; r < 8; ++r)
            acc[r] += __shfl_xor(acc[r], off, 32);
    }
    if (mn == 0) {
#pragma unroll
        for (int r = 0; r < 8; ++r)
            A_out[row_base + hl * 8 + r] = acc[r];
    }
}

// ---------------------------------------------------------------------------
// Kernel 3: per-segment max and exp-sum over the score vector (tiny: 1 MB).
// One block per segment; each block streams A + batch.
// ---------------------------------------------------------------------------
__global__ __launch_bounds__(256) void segstats_kernel(
    const float* __restrict__ A, const int* __restrict__ batch,
    float* __restrict__ seg_max, float* __restrict__ seg_sum)
{
    __shared__ float red[256];
    __shared__ float smax_sh;
    const int b = blockIdx.x;
    const int tid = threadIdx.x;

    float m = -3.402823466e38f;
    for (int i = tid; i < kN; i += 256)
        if (batch[i] == b) m = fmaxf(m, A[i]);
    red[tid] = m;
    __syncthreads();
    for (int s = 128; s > 0; s >>= 1) {
        if (tid < s) red[tid] = fmaxf(red[tid], red[tid + s]);
        __syncthreads();
    }
    if (tid == 0) { smax_sh = red[0]; seg_max[b] = red[0]; }
    __syncthreads();
    const float mx = smax_sh;

    float sum = 0.0f;
    for (int i = tid; i < kN; i += 256)
        if (batch[i] == b) sum += __expf(A[i] - mx);
    __syncthreads();
    red[tid] = sum;
    __syncthreads();
    for (int s = 128; s > 0; s >>= 1) {
        if (tid < s) red[tid] += red[tid + s];
        __syncthreads();
    }
    if (tid == 0) seg_sum[b] = red[0];
}

// ---------------------------------------------------------------------------
// Kernel 4: attn[i] = exp(A[i]-max[b])/sum[b]; slide[b,:] += attn[i]*x[i,:].
// Block owns 256 contiguous rows (segments are sorted -> at most a few
// boundary flushes); each thread owns 4 columns; one atomicAdd per
// (segment, column) per block.
// ---------------------------------------------------------------------------
__global__ __launch_bounds__(256) void aggregate_kernel(
    const float* __restrict__ x, const int* __restrict__ batch,
    const float* __restrict__ A,
    const float* __restrict__ seg_max, const float* __restrict__ seg_sum,
    float* __restrict__ slide, float* __restrict__ attn)
{
    __shared__ float s_attn[256];
    __shared__ int   s_b[256];
    const int tid = threadIdx.x;
    const int i0  = blockIdx.x * 256;
    const int i   = i0 + tid;

    const int b   = batch[i];
    const float e = __expf(A[i] - seg_max[b]) / seg_sum[b];
    s_attn[tid] = e;
    s_b[tid]    = b;
    attn[i]     = e;
    __syncthreads();

    const int d = tid * 4;
    float4 acc = make_float4(0.f, 0.f, 0.f, 0.f);
    int cur = s_b[0];
    for (int r = 0; r < 256; ++r) {
        const int rb = s_b[r];
        if (rb != cur) {
            atomicAdd(&slide[cur * kD + d + 0], acc.x);
            atomicAdd(&slide[cur * kD + d + 1], acc.y);
            atomicAdd(&slide[cur * kD + d + 2], acc.z);
            atomicAdd(&slide[cur * kD + d + 3], acc.w);
            acc = make_float4(0.f, 0.f, 0.f, 0.f);
            cur = rb;
        }
        const float4 xv = *(const float4*)(x + (size_t)(i0 + r) * kD + d);
        const float  av = s_attn[r];
        acc.x += av * xv.x;
        acc.y += av * xv.y;
        acc.z += av * xv.z;
        acc.w += av * xv.w;
    }
    atomicAdd(&slide[cur * kD + d + 0], acc.x);
    atomicAdd(&slide[cur * kD + d + 1], acc.y);
    atomicAdd(&slide[cur * kD + d + 2], acc.z);
    atomicAdd(&slide[cur * kD + d + 3], acc.w);
}

// ---------------------------------------------------------------------------
// Launch. Inputs (reference order): x, batch, Wv, bv, Wu, bu, w.
// d_out = [slide_features (16*1024) | attn (262144)] f32.
// ---------------------------------------------------------------------------
extern "C" void kernel_launch(void* const* d_in, const int* in_sizes, int n_in,
                              void* d_out, int out_size, void* d_ws, size_t ws_size,
                              hipStream_t stream)
{
    const float* x     = (const float*)d_in[0];
    const int*   batch = (const int*)d_in[1];
    const float* Wv    = (const float*)d_in[2];
    const float* bv    = (const float*)d_in[3];
    const float* Wu    = (const float*)d_in[4];
    const float* bu    = (const float*)d_in[5];
    const float* w     = (const float*)d_in[6];

    float* slide = (float*)d_out;
    float* attn  = slide + kB * kD;

    // Workspace layout: WvT bf16 | WuT bf16 | A f32 | seg_max | seg_sum
    bf16*  WvT = (bf16*)d_ws;
    bf16*  WuT = WvT + (size_t)kH * kD;
    float* A   = (float*)(WuT + (size_t)kH * kD);
    float* seg_max = A + kN;
    float* seg_sum = seg_max + kB;

    convert_kernel<<<(kH * kD + 255) / 256, 256, 0, stream>>>(Wv, Wu, WvT, WuT);
    zero_kernel<<<(kB * kD) / 256, 256, 0, stream>>>(slide);
    score_kernel<<<kN / (16 * 8), 256, 0, stream>>>(x, WvT, WuT, bv, bu, w, A);
    segstats_kernel<<<kB, 256, 0, stream>>>(A, batch, seg_max, seg_sum);
    aggregate_kernel<<<kN / 256, 256, 0, stream>>>(x, batch, A, seg_max, seg_sum,
                                                   slide, attn);
}
